// EncoderDecoder_13657996001769
// MI455X (gfx1250) — matrix operand order repaired and checked by reading, not compiled, not run on hardware
//
#include <hip/hip_runtime.h>
#include <cstdint>
#include <cstddef>

// ---------------------------------------------------------------------------
// Seq2seq LSTM encoder/decoder for MI455X (gfx1250), bf16 WMMA path.
//   - Weights pre-converted to bf16 and pre-swizzled into the native
//     v_wmma_f32_16x16x32_bf16 B-fragment layout (all fit in 192MB L2).
//   - Activations as bf16 [x | h] rows (double buffered), cell state f32.
//   - GEMM loops are software-pipelined: next k-step fragments are in
//     flight while the current WMMAs execute.
//   - k_logits computes both 16-row M tiles per wave so the dominant
//     W_out stream (64MB/step from L2) is fetched exactly once.
// ---------------------------------------------------------------------------

typedef __attribute__((ext_vector_type(16))) __bf16 v16bf;
typedef __attribute__((ext_vector_type(8)))  __bf16 v8bf;
typedef __attribute__((ext_vector_type(8)))  float  v8f;

#define EMBD   512
#define HIDD   1024
#define NB     32          // batch
#define SRCL   128
#define MAXL   128
#define VOC    32000
#define XHS    1536        // row stride of [x|h] activation buffer
#define KT_LSTM 48         // 1536 / 32
#define KT_OUT  32         // 1024 / 32
#define NT_LSTM 256        // 4096 / 16
#define NT_OUT  2000       // 32000 / 16

#define WMMA_BF16(a, b, c) \
  __builtin_amdgcn_wmma_f32_16x16x32_bf16(false, (a), false, (b), (short)0, (c), false, false)

// ---------------- bf16 helpers (round-to-nearest-even) ----------------------
__device__ __forceinline__ unsigned short f32_to_bf16(float f) {
  unsigned int u = __builtin_bit_cast(unsigned int, f);
  unsigned int r = u + 0x7FFFu + ((u >> 16) & 1u);
  return (unsigned short)(r >> 16);
}
__device__ __forceinline__ float bf16_to_f32(unsigned short h) {
  unsigned int u = ((unsigned int)h) << 16;
  return __builtin_bit_cast(float, u);
}

// B fragment: swizzled buffer stores, per 32x16 K-tile, 32 lanes x 16 bf16
// contiguously -> one aligned 32-byte load per lane.
__device__ __forceinline__ v16bf load_b_frag(const unsigned short* p, int lane) {
  return *(const v16bf*)(p + lane * 16);
}

// A fragment from a row-major bf16 activation row.
// Per ISA layout: lane holds row M=lane&15; lanes>=16 shift K by +8;
// elements 0..7 -> K = k0..k0+7, elements 8..15 -> K = k0+16..k0+23.
__device__ __forceinline__ v16bf load_a_frag(const unsigned short* base,
                                             int row, int k0) {
  const __bf16* q = (const __bf16*)(base + (size_t)row * XHS + k0);
  v8bf lo = *(const v8bf*)(q);
  v8bf hi = *(const v8bf*)(q + 16);
  v16bf v;
#pragma unroll
  for (int i = 0; i < 8; ++i) { v[i] = lo[i]; v[8 + i] = hi[i]; }
  return v;
}

__device__ __forceinline__ float sigmoidf_dev(float x) {
  return 1.0f / (1.0f + __expf(-x));
}

// ---------------------------------------------------------------------------
// Weight swizzle: rows [0, ksplit) from M0, rows [ksplit, K) from M1.
// Output layout: [nt][kt][lane*16 + e], element (lane,e) holds
//   src[K = kt*32 + (lane>=16)*8 + (e&7) + (e>=8)*16][N = nt*16 + (lane&15)]
// which is the native wmma_f32_16x16x32_bf16 B-operand layout.
// ---------------------------------------------------------------------------
__global__ void k_swizzleB(const float* __restrict__ M0,
                           const float* __restrict__ M1,
                           int ksplit, int ldn, int KT,
                           unsigned short* __restrict__ out, int total) {
  int idx = blockIdx.x * blockDim.x + threadIdx.x;
  if (idx >= total) return;
  int e    = idx & 15;
  int lane = (idx >> 4) & 31;
  int tile = idx >> 9;
  int kt   = tile % KT;
  int nt   = tile / KT;
  int K = kt * 32 + (e & 7) + ((e >= 8) ? 16 : 0) + ((lane >= 16) ? 8 : 0);
  int N = nt * 16 + (lane & 15);
  float v = (K < ksplit) ? M0[(size_t)K * ldn + N]
                         : M1[(size_t)(K - ksplit) * ldn + N];
  out[idx] = f32_to_bf16(v);
}

// Zero cell state and the h-part of activation buffer 0.
__global__ void k_init(float* __restrict__ c, unsigned short* __restrict__ xh0) {
  int idx = blockIdx.x * blockDim.x + threadIdx.x;
  if (idx >= NB * HIDD) return;
  c[idx] = 0.0f;
  int b = idx >> 10, k = idx & 1023;
  xh0[(size_t)b * XHS + EMBD + k] = 0;  // bf16 +0.0
}

// Gather encoder embedding for timestep t into x-part of xh.
__global__ void k_embed_enc(const int* __restrict__ inputs, int t,
                            const float* __restrict__ E,
                            unsigned short* __restrict__ xh) {
  int idx = blockIdx.x * blockDim.x + threadIdx.x;
  if (idx >= NB * EMBD) return;
  int b = idx >> 9, k = idx & 511;
  int tok = inputs[b * SRCL + t];
  xh[(size_t)b * XHS + k] = f32_to_bf16(E[(size_t)tok * EMBD + k]);
}

// Gather decoder embedding (tok0 == 1 at t == 0).
__global__ void k_embed_dec(const int* __restrict__ tokbuf, int t,
                            const float* __restrict__ E,
                            unsigned short* __restrict__ xh) {
  int idx = blockIdx.x * blockDim.x + threadIdx.x;
  if (idx >= NB * EMBD) return;
  int b = idx >> 9, k = idx & 511;
  int tok = (t == 0) ? 1 : tokbuf[b];
  xh[(size_t)b * XHS + k] = f32_to_bf16(E[(size_t)tok * EMBD + k]);
}

// ---------------------------------------------------------------------------
// One LSTM step. 128 waves: wave -> (mtile in {0,1}, ht in [0,64)).
// Each wave holds 4 gate accumulators (i,f,g,o) for the same 16x16 (M,N)
// block so the elementwise update is wave-local. The k-loop is software-
// pipelined: fragments for kt+1 are loaded before the kt WMMAs issue.
// Encoder passes `inputs` for masking; decoder passes nullptr.
// ---------------------------------------------------------------------------
__global__ void k_lstm(const unsigned short* __restrict__ xh_in,
                       unsigned short* __restrict__ xh_out,
                       float* __restrict__ c,
                       const unsigned short* __restrict__ Bsw,
                       const float* __restrict__ bias,
                       const int* __restrict__ inputs, int t) {
  int wid  = blockIdx.x * (blockDim.x >> 5) + (threadIdx.x >> 5);
  int lane = threadIdx.x & 31;
  int mtile = wid & 1;
  int ht    = wid >> 1;            // HID column tile, 0..63
  v8f acc0 = {}, acc1 = {}, acc2 = {}, acc3 = {};
  int arow = mtile * 16 + (lane & 15);
  int k0l  = (lane >= 16) ? 8 : 0;

  const unsigned short* bbase = Bsw + (size_t)ht * KT_LSTM * 512;
  const size_t gstride = (size_t)64 * KT_LSTM * 512;  // gate stride (elems)

  // Prologue: fragments for kt = 0.
  v16bf a  = load_a_frag(xh_in, arow, k0l);
  v16bf b0 = load_b_frag(bbase, lane);
  v16bf b1 = load_b_frag(bbase + gstride, lane);
  v16bf b2 = load_b_frag(bbase + 2 * gstride, lane);
  v16bf b3 = load_b_frag(bbase + 3 * gstride, lane);

  for (int kt = 0; kt + 1 < KT_LSTM; ++kt) {
    const unsigned short* bp = bbase + (size_t)(kt + 1) * 512;
    v16bf an  = load_a_frag(xh_in, arow, (kt + 1) * 32 + k0l);
    v16bf b0n = load_b_frag(bp, lane);
    v16bf b1n = load_b_frag(bp + gstride, lane);
    v16bf b2n = load_b_frag(bp + 2 * gstride, lane);
    v16bf b3n = load_b_frag(bp + 3 * gstride, lane);
    acc0 = WMMA_BF16(a, b0, acc0);
    acc1 = WMMA_BF16(a, b1, acc1);
    acc2 = WMMA_BF16(a, b2, acc2);
    acc3 = WMMA_BF16(a, b3, acc3);
    a = an; b0 = b0n; b1 = b1n; b2 = b2n; b3 = b3n;
  }
  acc0 = WMMA_BF16(a, b0, acc0);
  acc1 = WMMA_BF16(a, b1, acc1);
  acc2 = WMMA_BF16(a, b2, acc2);
  acc3 = WMMA_BF16(a, b3, acc3);

  int ncol  = ht * 16 + (lane & 15);        // HID column
  int mhalf = (lane >= 16) ? 8 : 0;
#pragma unroll
  for (int r = 0; r < 8; ++r) {
    int brow = mtile * 16 + r + mhalf;      // batch row
    float iv = sigmoidf_dev(acc0[r] + bias[ncol]);
    float fv = sigmoidf_dev(acc1[r] + bias[HIDD + ncol]);
    float gv = tanhf(acc2[r] + bias[2 * HIDD + ncol]);
    float ov = sigmoidf_dev(acc3[r] + bias[3 * HIDD + ncol]);
    float cold = c[(size_t)brow * HIDD + ncol];
    float cn = fv * cold + iv * gv;
    float hn = ov * tanhf(cn);
    if (inputs) {  // encoder padding mask
      bool m = inputs[brow * SRCL + t] > 0;
      if (!m) {
        cn = cold;
        hn = bf16_to_f32(xh_in[(size_t)brow * XHS + EMBD + ncol]);
      }
    }
    c[(size_t)brow * HIDD + ncol] = cn;
    xh_out[(size_t)brow * XHS + EMBD + ncol] = f32_to_bf16(hn);
  }
}

// ---------------------------------------------------------------------------
// Decoder logits: h @ W_out + b_out. 2000 waves; each wave owns one
// 16-column vocab tile and BOTH 16-row M tiles, so every W_out fragment
// is fetched from L2 exactly once and feeds two WMMAs. Software-pipelined
// with a 2-tiles-ahead prefetch. Writes straight into d_out[(b,t,:)].
// ---------------------------------------------------------------------------
__global__ void k_logits(const unsigned short* __restrict__ xh,   // h in [512,1536)
                         const unsigned short* __restrict__ BswOut,
                         const float* __restrict__ b_out,
                         float* __restrict__ out, int t) {
  int wid  = blockIdx.x * (blockDim.x >> 5) + (threadIdx.x >> 5);
  int lane = threadIdx.x & 31;
  int nt   = wid;                  // 0..1999 (uniform per wave)
  v8f acc0 = {}, acc1 = {};
  const unsigned short* xh_h = xh + EMBD;   // h part, same row stride
  int ar0 = (lane & 15);           // M tile 0 rows
  int ar1 = 16 + (lane & 15);      // M tile 1 rows
  int k0l = (lane >= 16) ? 8 : 0;

  const unsigned short* bbase = BswOut + (size_t)nt * KT_OUT * 512;

  // Prologue: fragments for kt = 0.
  v16bf a0 = load_a_frag(xh_h, ar0, k0l);
  v16bf a1 = load_a_frag(xh_h, ar1, k0l);
  v16bf b  = load_b_frag(bbase, lane);

  for (int kt = 0; kt + 1 < KT_OUT; ++kt) {
    int k0 = (kt + 1) * 32 + k0l;
    __builtin_prefetch((const void*)(bbase + (size_t)(kt + 2) * 512), 0, 0);
    v16bf a0n = load_a_frag(xh_h, ar0, k0);
    v16bf a1n = load_a_frag(xh_h, ar1, k0);
    v16bf bn  = load_b_frag(bbase + (size_t)(kt + 1) * 512, lane);
    acc0 = WMMA_BF16(a0, b, acc0);
    acc1 = WMMA_BF16(a1, b, acc1);
    a0 = a0n; a1 = a1n; b = bn;
  }
  acc0 = WMMA_BF16(a0, b, acc0);
  acc1 = WMMA_BF16(a1, b, acc1);

  int n  = nt * 16 + (lane & 15);
  float bo = b_out[n];
  int mhalf = (lane >= 16) ? 8 : 0;
#pragma unroll
  for (int r = 0; r < 8; ++r) {
    int brow0 = r + mhalf;           // M tile 0 batch row
    int brow1 = 16 + r + mhalf;      // M tile 1 batch row
    out[((size_t)brow0 * MAXL + t) * VOC + n] = acc0[r] + bo;
    out[((size_t)brow1 * MAXL + t) * VOC + n] = acc1[r] + bo;
  }
}

// Greedy argmax over one row of logits (first-max tie break, like jnp.argmax).
__global__ void k_argmax(const float* __restrict__ out, int t,
                         int* __restrict__ tokbuf, float* __restrict__ tokout) {
  int b = blockIdx.x;
  const float* row = out + ((size_t)b * MAXL + t) * VOC;
  float best = -3.402823466e+38f;
  int bidx = 0;
  for (int n = threadIdx.x; n < VOC; n += blockDim.x) {
    float v = row[n];
    if (v > best) { best = v; bidx = n; }
  }
  __shared__ float sv[256];
  __shared__ int   si[256];
  sv[threadIdx.x] = best; si[threadIdx.x] = bidx;
  __syncthreads();
  for (int s = 128; s > 0; s >>= 1) {
    if ((int)threadIdx.x < s) {
      float v2 = sv[threadIdx.x + s]; int i2 = si[threadIdx.x + s];
      if (v2 > sv[threadIdx.x] ||
          (v2 == sv[threadIdx.x] && i2 < si[threadIdx.x])) {
        sv[threadIdx.x] = v2; si[threadIdx.x] = i2;
      }
    }
    __syncthreads();
  }
  if (threadIdx.x == 0) {
    tokbuf[b] = si[0];
    tokout[b * MAXL + t] = (float)si[0];
  }
}

// ---------------------------------------------------------------------------
extern "C" void kernel_launch(void* const* d_in, const int* in_sizes, int n_in,
                              void* d_out, int out_size, void* d_ws, size_t ws_size,
                              hipStream_t stream) {
  const int*   inputs = (const int*)d_in[0];
  const float* E_enc  = (const float*)d_in[1];
  const float* W_enc  = (const float*)d_in[2];
  const float* U_enc  = (const float*)d_in[3];
  const float* b_enc  = (const float*)d_in[4];
  const float* E_dec  = (const float*)d_in[5];
  const float* W_dec  = (const float*)d_in[6];
  const float* U_dec  = (const float*)d_in[7];
  const float* b_dec  = (const float*)d_in[8];
  const float* W_out  = (const float*)d_in[9];
  const float* b_out  = (const float*)d_in[10];
  float* out = (float*)d_out;
  float* tokout = out + (size_t)NB * MAXL * VOC;

  // Workspace carve-up (bf16 weights + activations + f32 cell state).
  const size_t nBL = (size_t)NT_LSTM * KT_LSTM * 512;  // 6,291,456 elems
  const size_t nBO = (size_t)NT_OUT  * KT_OUT  * 512;  // 32,768,000 elems
  const size_t nXH = (size_t)NB * XHS;                 // 49,152 elems
  unsigned short* BswEnc = (unsigned short*)d_ws;
  unsigned short* BswDec = BswEnc + nBL;
  unsigned short* BswOut = BswDec + nBL;
  unsigned short* xh0    = BswOut + nBO;
  unsigned short* xh1    = xh0 + nXH;
  float* cbuf  = (float*)(xh1 + nXH);
  int*   tokbf = (int*)(cbuf + (size_t)NB * HIDD);

  // One-time (per call) weight conversion + swizzle to bf16 B-fragments.
  k_swizzleB<<<(int)(nBL / 256), 256, 0, stream>>>(W_enc, U_enc, EMBD, 4 * HIDD, KT_LSTM, BswEnc, (int)nBL);
  k_swizzleB<<<(int)(nBL / 256), 256, 0, stream>>>(W_dec, U_dec, EMBD, 4 * HIDD, KT_LSTM, BswDec, (int)nBL);
  k_swizzleB<<<(int)(nBO / 256), 256, 0, stream>>>(W_out, W_out, HIDD, VOC, KT_OUT, BswOut, (int)nBO);
  k_init<<<(NB * HIDD) / 256, 256, 0, stream>>>(cbuf, xh0);

  // Encoder: 128 masked LSTM steps, h double-buffered in [x|h] bf16 rows.
  for (int t = 0; t < SRCL; ++t) {
    unsigned short* cur = (t & 1) ? xh1 : xh0;
    unsigned short* nxt = (t & 1) ? xh0 : xh1;
    k_embed_enc<<<(NB * EMBD) / 256, 256, 0, stream>>>(inputs, t, E_enc, cur);
    k_lstm<<<16, 256, 0, stream>>>(cur, nxt, cbuf, BswEnc, b_enc, inputs, t);
  }

  // Decoder: greedy autoregressive loop; final encoder h lands in xh0.
  for (int t = 0; t < MAXL; ++t) {
    unsigned short* cur = (t & 1) ? xh1 : xh0;
    unsigned short* nxt = (t & 1) ? xh0 : xh1;
    k_embed_dec<<<(NB * EMBD) / 256, 256, 0, stream>>>(tokbf, t, E_dec, cur);
    k_lstm<<<16, 256, 0, stream>>>(cur, nxt, cbuf, BswDec, b_dec, nullptr, t);
    k_logits<<<NT_OUT / 8, 256, 0, stream>>>(nxt, BswOut, b_out, out, t);
    k_argmax<<<NB, 256, 0, stream>>>(out, t, tokbf, tokout);
  }
}
